// NodeWiseMamba_8546984919674
// MI455X (gfx1250) — compile-verified, matrix operand order checked by reading, not run on hardware
//
#include <hip/hip_runtime.h>
#include <math.h>

// ---------------- types ----------------
typedef __attribute__((ext_vector_type(16))) __bf16 v16bf;
typedef __attribute__((ext_vector_type(8)))  float  v8f;

#define NT      4096
#define TT      24
#define HID     256
#define DIN     256
#define DSTATE  16
#define DTRANK  16
#define NGROUP  2                 // nodes per workgroup
#define ROWS    (NGROUP * TT)     // 48 rows per workgroup
#define LDSB    272               // bf16 LDS row stride (x16 elems -> 32B aligned, 8-bank skew)

// ---- Aliased LDS pool (lifetimes are phase-disjoint, barriers between) ----
// [0,      49152)  zg (st1-5)            | oy (st6-9)
// [49152,  75264)  xb (st0-1)            | xcv (st2-5) | sc/wgt/pooled (st7-9)
// [75264, 124416)  f32 staging (st0)     | xm (st1-2)  | dt (st4-5)
// [124416,133632)  xdbl (st3-5)
// [133632,159744)  yb (st5-6)
#define OFF_ZG     0
#define OFF_OY     0
#define OFF_XB     49152
#define OFF_XCV    49152
#define OFF_SC     49152
#define OFF_WGT    (49152 + 256)
#define OFF_POOLED (49152 + 512)
#define OFF_STAGE  75264
#define OFF_XM     75264
#define OFF_DT     75264
#define OFF_XDBL   124416
#define OFF_YB     133632
#define MEGA_LDS   159744         // 156 KB -> 2 workgroups (16 waves) per 320KB WGP

// ---- WMMA helper: D = A(16x32 bf16) * B(32x16 bf16) + C(f32) ----
__device__ __forceinline__ v8f wmma_bf16(v16bf a, v16bf b, v8f c) {
  return __builtin_amdgcn_wmma_f32_16x16x32_bf16(false, a, false, b, (short)0, c,
                                                 false, false);
}

// Fragment load (contiguous k order, same for A and B): lanes 0-15 hold row/col t0+lane with K=k0..k0+15,
// lanes 16-31 hold row/col t0+lane-16 with K=k0+16..k0+31.
__device__ __forceinline__ v16bf load_frag(const __bf16* base, int stride,
                                           int t0, int k0, int lane) {
  const __bf16* p = base + (t0 + (lane & 15)) * stride + k0 + ((lane >> 4) << 4);
  return *(const v16bf*)p;   // 32B: 2x b128 (32B-aligned by construction)
}

__device__ __forceinline__ float silu_f(float x) {
  return x / (1.0f + __expf(-x));
}

// CDNA5 async copy: global -> LDS, 16B per lane, tracked by ASYNCcnt.
__device__ __forceinline__ void async_g2l_b128(unsigned lds_byte_addr,
                                               unsigned long long gaddr) {
  asm volatile("global_load_async_to_lds_b128 %0, %1, off"
               :: "v"(lds_byte_addr), "v"(gaddr) : "memory");
}
__device__ __forceinline__ void wait_asynccnt0() {
  asm volatile("s_wait_asynccnt 0x0" ::: "memory");
}

// ================= prep kernel 1: Wfused = Wi @ Win (store N-major bf16) =====
__global__ __launch_bounds__(256) void prep_fuse(const float* __restrict__ Wi,
                                                 const float* __restrict__ bi,
                                                 const float* __restrict__ Win,
                                                 __bf16* __restrict__ WfT,
                                                 float* __restrict__ bfused) {
  const int n = blockIdx.x;      // 0..511 (output column of Win)
  const int i = threadIdx.x;     // 0..255 (input dim)
  float acc = 0.0f;
  for (int h = 0; h < HID; ++h)
    acc += Wi[i * HID + h] * Win[h * 512 + n];
  WfT[n * 256 + i] = (__bf16)acc;
  if (i == 0) {
    float b = 0.0f;
    for (int h = 0; h < HID; ++h) b += bi[h] * Win[h * 512 + n];
    bfused[n] = b;
  }
}

// ====== prep kernel 2: transpose/convert Wx, Wssm_out; A = -exp(A_log) ======
__global__ __launch_bounds__(256) void prep_misc(const float* __restrict__ Wx,
                                                 const float* __restrict__ Wssm,
                                                 const float* __restrict__ A_log,
                                                 __bf16* __restrict__ WxT,
                                                 __bf16* __restrict__ WoutT,
                                                 float* __restrict__ Aws) {
  const int idx = blockIdx.x * 256 + threadIdx.x;
  const int NWX = 48 * 256, NWO = 256 * 256, NA = 256 * 16;
  if (idx < NWX) {
    int n = idx >> 8, k = idx & 255;
    WxT[idx] = (__bf16)Wx[k * 48 + n];
  } else if (idx < NWX + NWO) {
    int j = idx - NWX;
    int n = j >> 8, k = j & 255;
    WoutT[j] = (__bf16)Wssm[k * 256 + n];
  } else if (idx < NWX + NWO + NA) {
    int a = idx - (NWX + NWO);
    Aws[a] = -__expf(A_log[a]);
  }
}

// ========================== fused megakernel ==============================
// One 256-thread (8 wave32) workgroup processes NGROUP=2 nodes end-to-end.
__global__ __launch_bounds__(256) void mamba_mega(
    const float* __restrict__ x_hist,
    const float* __restrict__ convW, const float* __restrict__ convb,
    const float* __restrict__ Wdt,   const float* __restrict__ bdt,
    const float* __restrict__ Dp,
    const float* __restrict__ Wattn, const float* __restrict__ battn,
    const float* __restrict__ Wo,    const float* __restrict__ bo,
    const __bf16* __restrict__ WfT,  const float* __restrict__ bfused,
    const __bf16* __restrict__ WxT,  const __bf16* __restrict__ WoutT,
    const float* __restrict__ Aws,
    float* __restrict__ out) {
  extern __shared__ __attribute__((aligned(64))) char smem[];
  float*  zg   = (float*)(smem + OFF_ZG);      // [48][256] gate
  float*  oy   = (float*)(smem + OFF_OY);      // [48][256] out-proj (aliases zg)
  __bf16* xb   = (__bf16*)(smem + OFF_XB);     // [48][272] input bf16
  __bf16* xcv  = (__bf16*)(smem + OFF_XCV);    // [48][272] conv+silu (aliases xb)
  float*  stg  = (float*)(smem + OFF_STAGE);   // [48][256] f32 async staging
  float*  xm   = (float*)(smem + OFF_XM);      // [48][256] ssm stream (aliases stg)
  float*  dt   = (float*)(smem + OFF_DT);      // [48][256] softplus(dt) (aliases xm)
  float*  xdbl = (float*)(smem + OFF_XDBL);    // [48][48]  dt_raw|B|C
  __bf16* yb   = (__bf16*)(smem + OFF_YB);     // [48][272] gated y
  float*  sc   = (float*)(smem + OFF_SC);      // [48] attn scores
  float*  wgt  = (float*)(smem + OFF_WGT);     // [48] softmax weights
  float*  pooled = (float*)(smem + OFF_POOLED);// [2][256]

  const int tid  = threadIdx.x;
  const int lane = tid & 31;
  const int wave = tid >> 5;
  const int node0 = blockIdx.x * NGROUP;

  if (tid == 0) {
    __builtin_prefetch(WfT, 0, 1);             // global_prefetch_b8 (weights -> L2)
    __builtin_prefetch(WoutT, 0, 1);
  }

  // ---- stage 0a: async-copy 48x256 f32 x_hist block straight into LDS -----
  {
    const char* gbase = (const char*)(x_hist + (size_t)node0 * TT * 256);
    const unsigned lbase = (unsigned)(size_t)(void*)stg;
    for (int i = 0; i < 12; ++i) {             // 3072 chunks * 16B = 48KB
      const unsigned c16 = (unsigned)(tid + i * 256);
      async_g2l_b128(lbase + c16 * 16u,
                     (unsigned long long)(size_t)(gbase + (size_t)c16 * 16u));
    }
    wait_asynccnt0();
  }
  __syncthreads();

  // ---- stage 0b: convert staged f32 -> bf16 fragment buffer ---------------
  {
    const float4* s4 = (const float4*)stg;
    for (int i = 0; i < 12; ++i) {
      int e4 = tid + i * 256;
      float4 v = s4[e4];
      int e = e4 * 4;
      int row = e >> 8, col = e & 255;
      __bf16* d = xb + row * LDSB + col;
      d[0] = (__bf16)v.x; d[1] = (__bf16)v.y;
      d[2] = (__bf16)v.z; d[3] = (__bf16)v.w;
    }
  }
  __syncthreads();

  // ---- stage 1: xz[48,512] = X @ Wfused + bfused  (96 tiles, WMMA) --------
  for (int tile = wave; tile < 96; tile += 8) {
    const int m0 = (tile % 3) * 16;
    const int n0 = (tile / 3) * 16;
    v8f c = {0.f, 0.f, 0.f, 0.f, 0.f, 0.f, 0.f, 0.f};
    for (int k0 = 0; k0 < 256; k0 += 32) {
      v16bf a = load_frag(xb, LDSB, m0, k0, lane);
      v16bf b = load_frag(WfT, 256, n0, k0, lane);
      c = wmma_bf16(a, b, c);
    }
    const int n = n0 + (lane & 15);
    const float bf = bfused[n];
    const int mbase = m0 + ((lane >> 4) << 3);
    float* dstp = (n0 < 256) ? xm : zg;
    const int ncol = (n0 < 256) ? n : (n - 256);
    for (int r = 0; r < 8; ++r)
      dstp[(mbase + r) * 256 + ncol] = c[r] + bf;
  }
  __syncthreads();

  // ---- stage 2: causal depthwise conv (k=4) + bias + SiLU -> bf16 ---------
  for (int g = 0; g < NGROUP; ++g) {
    const int d = tid;
    const float w0 = convW[d * 4 + 0], w1 = convW[d * 4 + 1];
    const float w2 = convW[d * 4 + 2], w3 = convW[d * 4 + 3];
    const float bc = convb[d];
    float xq0 = 0.f, xq1 = 0.f, xq2 = 0.f;
    const int rbase = g * TT;
    for (int t = 0; t < TT; ++t) {
      float xc = xm[(rbase + t) * 256 + d];
      float a = w0 * xq0 + w1 * xq1 + w2 * xq2 + w3 * xc + bc;
      xcv[(rbase + t) * LDSB + d] = (__bf16)silu_f(a);
      xq0 = xq1; xq1 = xq2; xq2 = xc;
    }
  }
  __syncthreads();

  // ---- stage 3: xdbl[48,48] = xconv @ Wx  (9 tiles, WMMA) -----------------
  for (int tile = wave; tile < 9; tile += 8) {
    const int m0 = (tile % 3) * 16;
    const int n0 = (tile / 3) * 16;
    v8f c = {0.f, 0.f, 0.f, 0.f, 0.f, 0.f, 0.f, 0.f};
    for (int k0 = 0; k0 < 256; k0 += 32) {
      v16bf a = load_frag(xcv, LDSB, m0, k0, lane);
      v16bf b = load_frag(WxT, 256, n0, k0, lane);
      c = wmma_bf16(a, b, c);
    }
    const int n = n0 + (lane & 15);
    const int mbase = m0 + ((lane >> 4) << 3);
    for (int r = 0; r < 8; ++r)
      xdbl[(mbase + r) * 48 + n] = c[r];
  }
  __syncthreads();

  // ---- stage 4: dt = softplus(dt_raw @ Wdt + bdt) -------------------------
  {
    const int d = tid;
    float wj[DTRANK];
    for (int j = 0; j < DTRANK; ++j) wj[j] = Wdt[j * 256 + d];
    const float bb = bdt[d];
    for (int row = 0; row < ROWS; ++row) {
      float acc = bb;
      for (int j = 0; j < DTRANK; ++j) acc += xdbl[row * 48 + j] * wj[j];
      dt[row * 256 + d] = (acc > 20.f) ? acc : log1pf(__expf(acc));
    }
  }
  __syncthreads();

  // ---- stage 5: selective-scan over T; fuse skip, gate -> bf16 ------------
  {
    const int d = tid;
    float Ad[DSTATE];
    for (int j = 0; j < DSTATE; ++j) Ad[j] = Aws[d * DSTATE + j];
    const float dp = Dp[d];
    for (int g = 0; g < NGROUP; ++g) {
      float h[DSTATE];
      for (int j = 0; j < DSTATE; ++j) h[j] = 0.f;
      for (int t = 0; t < TT; ++t) {
        const int row = g * TT + t;
        const float dtv = dt[row * 256 + d];
        const float xv  = (float)xcv[row * LDSB + d];
        const float zv  = zg[row * 256 + d];
        const float prod = dtv * xv;
        float y = 0.f;
        for (int j = 0; j < DSTATE; ++j) {
          const float dA = __expf(dtv * Ad[j]);
          const float Bv = xdbl[row * 48 + DTRANK + j];
          const float Cv = xdbl[row * 48 + DTRANK + DSTATE + j];
          h[j] = dA * h[j] + prod * Bv;
          y += h[j] * Cv;
        }
        const float yp = (y + xv * dp) * silu_f(zv);
        yb[row * LDSB + d] = (__bf16)yp;
      }
    }
  }
  __syncthreads();

  // ---- stage 6: oy[48,256] = yb @ Wssm_out  (48 tiles, WMMA) --------------
  // (oy aliases zg: zg is dead after stage 5)
  for (int tile = wave; tile < 48; tile += 8) {
    const int m0 = (tile % 3) * 16;
    const int n0 = (tile / 3) * 16;
    v8f c = {0.f, 0.f, 0.f, 0.f, 0.f, 0.f, 0.f, 0.f};
    for (int k0 = 0; k0 < 256; k0 += 32) {
      v16bf a = load_frag(yb, LDSB, m0, k0, lane);
      v16bf b = load_frag(WoutT, 256, n0, k0, lane);
      c = wmma_bf16(a, b, c);
    }
    const int n = n0 + (lane & 15);
    const int mbase = m0 + ((lane >> 4) << 3);
    for (int r = 0; r < 8; ++r)
      oy[(mbase + r) * 256 + n] = c[r];
  }
  __syncthreads();

  // ---- stage 7: attention scores + softmax over T -------------------------
  if (tid < ROWS) {
    float acc = battn[0];
    for (int d2 = 0; d2 < 256; ++d2) acc += oy[tid * 256 + d2] * Wattn[d2];
    sc[tid] = acc;
  }
  __syncthreads();
  if (tid < NGROUP) {
    float m = -1e30f;
    for (int t = 0; t < TT; ++t) m = fmaxf(m, sc[tid * TT + t]);
    float s = 0.f;
    for (int t = 0; t < TT; ++t) {
      float e = __expf(sc[tid * TT + t] - m);
      wgt[tid * TT + t] = e; s += e;
    }
    const float inv = 1.f / s;
    for (int t = 0; t < TT; ++t) wgt[tid * TT + t] *= inv;
  }
  __syncthreads();

  // ---- stage 8: pooled = sum_t w_t * oy_t ---------------------------------
  for (int g = 0; g < NGROUP; ++g) {
    const int d = tid;
    float acc = 0.f;
    for (int t = 0; t < TT; ++t)
      acc += wgt[g * TT + t] * oy[(g * TT + t) * 256 + d];
    pooled[g * 256 + d] = acc;
  }
  __syncthreads();

  // ---- stage 9: out = GELU(pooled @ Wo + bo) ------------------------------
  for (int g = 0; g < NGROUP; ++g) {
    const int d = tid;
    float acc = bo[d];
    for (int k = 0; k < 256; ++k) acc += pooled[g * 256 + k] * Wo[k * 256 + d];
    const float ge = 0.5f * acc * (1.0f + erff(acc * 0.70710678118654752f));
    out[(size_t)(node0 + g) * 256 + d] = ge;
  }
}

// ============================ host launcher ================================
extern "C" void kernel_launch(void* const* d_in, const int* in_sizes, int n_in,
                              void* d_out, int out_size, void* d_ws, size_t ws_size,
                              hipStream_t stream) {
  const float* x_hist = (const float*)d_in[0];
  const float* Wi     = (const float*)d_in[1];
  const float* bi     = (const float*)d_in[2];
  const float* Win    = (const float*)d_in[3];
  const float* convW  = (const float*)d_in[4];
  const float* convb  = (const float*)d_in[5];
  const float* Wx     = (const float*)d_in[6];
  const float* Wdt    = (const float*)d_in[7];
  const float* bdt    = (const float*)d_in[8];
  const float* A_log  = (const float*)d_in[9];
  const float* Dp     = (const float*)d_in[10];
  const float* Wssm   = (const float*)d_in[11];
  const float* Wattn  = (const float*)d_in[12];
  const float* battn  = (const float*)d_in[13];
  const float* Wo     = (const float*)d_in[14];
  const float* bo     = (const float*)d_in[15];

  char* ws = (char*)d_ws;
  __bf16* WfT    = (__bf16*)(ws + 0);        // 512*256 bf16 = 262144 B
  float*  bfused = (float*)(ws + 262144);    // 512 f32    =   2048 B
  __bf16* WxT    = (__bf16*)(ws + 264192);   // 48*256 bf16 =  24576 B
  __bf16* WoutT  = (__bf16*)(ws + 288768);   // 256*256 bf16= 131072 B
  float*  Aws    = (float*)(ws + 419840);    // 256*16 f32  =  16384 B

  prep_fuse<<<512, 256, 0, stream>>>(Wi, bi, Win, WfT, bfused);
  prep_misc<<<320, 256, 0, stream>>>(Wx, Wssm, A_log, WxT, WoutT, Aws);

  (void)hipFuncSetAttribute((const void*)mamba_mega,
                            hipFuncAttributeMaxDynamicSharedMemorySize,
                            (int)MEGA_LDS);

  mamba_mega<<<NT / NGROUP, 256, MEGA_LDS, stream>>>(
      x_hist, convW, convb, Wdt, bdt, Dp, Wattn, battn, Wo, bo,
      WfT, bfused, WxT, WoutT, Aws, (float*)d_out);
}
